// TemporalTruthModel_31413390803233
// MI455X (gfx1250) — compile-verified, matrix operand order checked by reading, not run on hardware
//
#include <hip/hip_runtime.h>

typedef unsigned short u16;
typedef __attribute__((ext_vector_type(16))) __bf16 v16bf;
typedef __attribute__((ext_vector_type(8)))  float  v8f;

union FragU { uint4 q[2]; v16bf v; };

#define GSTR 388   // padded f32 row stride for gate LDS tiles (388 % 64 = 4 banks)

__device__ __forceinline__ unsigned f2bf_u(float f) {
  unsigned u = __float_as_uint(f);
  return ((u + 0x7FFFu + ((u >> 16) & 1u)) >> 16) & 0xFFFFu;
}
__device__ __forceinline__ float bf2f(unsigned s) { return __uint_as_float(s << 16); }

__device__ __forceinline__ uint4 pack8(const float v[8]) {
  uint4 q;
  q.x = f2bf_u(v[0]) | (f2bf_u(v[1]) << 16);
  q.y = f2bf_u(v[2]) | (f2bf_u(v[3]) << 16);
  q.z = f2bf_u(v[4]) | (f2bf_u(v[5]) << 16);
  q.w = f2bf_u(v[6]) | (f2bf_u(v[7]) << 16);
  return q;
}

// A/B fragment element order (wave32, 16-bit, 16x32): lane L, elem e ->
// row = L%16, K = 8*(L>=16) + (e<8 ? e : e+8).  pack_off inverts (m,k).
__device__ __forceinline__ int pack_off(int m, int k) {
  int kt = k >> 5, grp = (k >> 3) & 3;
  int lane = (m & 15) + ((grp & 1) << 4);
  int eb = (grp >> 1) << 3;
  return (kt * 32 + lane) * 16 + eb;
}

__device__ __forceinline__ v8f wmma_bf16(const FragU& A, const FragU& B, v8f C) {
  return __builtin_amdgcn_wmma_f32_16x16x32_bf16(false, A.v, false, B.v, (short)0, C,
                                                 false, false);
}

__device__ __forceinline__ float sigf(float x) { return 1.0f / (1.0f + __expf(-x)); }
__device__ __forceinline__ float tanhf_fast(float x) {
  float e = __expf(-2.0f * x);
  return (1.0f - e) / (1.0f + e);
}

// ---------------------------------------------------------------------------
// Weight packer: W[o*stride + k] (fp32) -> bf16 B-fragment order
// ---------------------------------------------------------------------------
__global__ __launch_bounds__(256) void pack_w_kernel(const float* __restrict__ W,
                                                     u16* __restrict__ out, int O, int K,
                                                     int stride) {
  int blk = blockIdx.x * 256 + threadIdx.x;
  int nb = K >> 3;
  if (blk >= O * nb) return;
  int o = blk / nb, k0 = (blk % nb) << 3;
  int KT = K >> 5;
  int kt = k0 >> 5, grp = (k0 >> 3) & 3;
  int lane = (o & 15) + ((grp & 1) << 4);
  int eb = (grp >> 1) << 3;
  float v[8];
#pragma unroll
  for (int i = 0; i < 8; ++i) v[i] = W[(size_t)o * stride + k0 + i];
  *(uint4*)&out[(((size_t)((o >> 4) * KT + kt)) * 32 + lane) * 16 + eb] = pack8(v);
}

__global__ __launch_bounds__(256) void copy_wy_kernel(const float* __restrict__ W,
                                                      float* __restrict__ wy) {
  int i = blockIdx.x * 256 + threadIdx.x;
  if (i < 384) wy[i] = W[(size_t)i * 129 + 128];
}

__global__ __launch_bounds__(256) void zero_f32_kernel(float* p, int n) {
  int i = blockIdx.x * 256 + threadIdx.x;
  if (i < n) p[i] = 0.0f;
}
__global__ __launch_bounds__(256) void zero_u16_kernel(u16* p, int n) {
  int i = blockIdx.x * 256 + threadIdx.x;
  if (i < n) p[i] = 0;
}

__global__ __launch_bounds__(256) void deg_kernel(const int* __restrict__ src,
                                                  float* __restrict__ deg, int n) {
  int i = blockIdx.x * 256 + threadIdx.x;
  if (i < n) atomicAdd(&deg[src[i]], 1.0f);
}

__global__ __launch_bounds__(256) void scatter_kernel(const int* __restrict__ src,
                                                      const int* __restrict__ dst,
                                                      const float* __restrict__ hg,
                                                      float* __restrict__ acc, int nE) {
  int i = blockIdx.x * 256 + threadIdx.x;
  int e = i >> 7, d = i & 127;
  if (e < nE)
    atomicAdd(&acc[(size_t)src[e] * 128 + d], hg[(size_t)dst[e] * 128 + d]);
}

__global__ __launch_bounds__(256) void copy_truth_kernel(const float* __restrict__ t,
                                                         float* __restrict__ o, int n) {
  int i = blockIdx.x * 256 + threadIdx.x;
  if (i < n) o[i] = t[i];
}

// ---------------------------------------------------------------------------
// Stage 1: comment GRU.  Grid 4096 groups x 256 thr; group = 16 rows of
// B = T*N, full 8-step GRU.  gi: 16x384x256 bf16 WMMA, gh: 16x384x128.
// The next step's fp32 comment tile is prefetched into a double-buffered LDS
// staging area with GLOBAL_LOAD_ASYNC_TO_LDS (ASYNCcnt) while the current
// step's WMMAs run.  Weight fragments are re-read from global memory each step.
// ---------------------------------------------------------------------------
__global__ __launch_bounds__(256, 1) void comment_gru_kernel(
    const float* __restrict__ comments, const u16* __restrict__ W1p,
    const u16* __restrict__ W2p, const float* __restrict__ bih,
    const float* __restrict__ bhh, u16* __restrict__ hc_pack) {
  __shared__ __align__(16) float s_xf[2][16 * 256];  // fp32 async staging (32 KB)
  __shared__ __align__(16) u16 s_xp[8 * 512];        // x tile, A-frag order (K=256)
  __shared__ __align__(16) u16 s_hp[4 * 512];        // h tile, A-frag order (K=128)
  __shared__ float s_gi[16 * GSTR];
  __shared__ float s_gh[16 * GSTR];
  __shared__ float s_bi[384], s_bh[384];

  const int tid = threadIdx.x, lane = tid & 31, wv = tid >> 5;
  const int g = blockIdx.x;
  const int t = g >> 8, ntile = g & 255;
  const int mrow = (lane >> 4) << 3, col0 = lane & 15;

  for (int i = tid; i < 4 * 512; i += 256) s_hp[i] = 0;
  for (int i = tid; i < 384; i += 256) { s_bi[i] = bih[i]; s_bh[i] = bhh[i]; }

  const float* base = comments + ((size_t)(t * 4096 + ntile * 16)) * 8 * 256;
  const unsigned lds_xf = (unsigned)(size_t)&s_xf[0][0];  // LDS byte offset (low 32b)

  // Async-DMA one 16x1KB comment tile (strided rows) into staging buffer `buf`.
  auto issue_tile = [&](int cc, int buf) {
#pragma unroll
    for (int k = 0; k < 4; ++k) {
      int chunk = tid + 256 * k;              // 1024 x 16B chunks
      int r = chunk >> 6, c16 = chunk & 63;   // row, 16B-chunk within row
      const float* gp = base + ((size_t)r * 8 + cc) * 256 + c16 * 4;
      unsigned la = lds_xf + (unsigned)(buf * 16384 + chunk * 16);
      asm volatile("global_load_async_to_lds_b128 %0, %1, off"
                   :
                   : "v"(la), "v"((unsigned long long)(size_t)gp)
                   : "memory");
    }
  };

  issue_tile(0, 0);

  for (int c = 0; c < 8; ++c) {
    const int buf = c & 1;
    // prefetch next tile into the other buffer ((c+1)&7 wraps harmlessly at c=7)
    issue_tile((c + 1) & 7, buf ^ 1);
    // oldest 4 async loads (current buffer) must be complete
    asm volatile("s_wait_asynccnt 0x4" ::: "memory");
    __syncthreads();  // all waves' DMA for `buf` visible in LDS

    // ---- convert staged fp32 tile -> bf16 A-fragment pack ------------------
#pragma unroll
    for (int i = 0; i < 2; ++i) {
      int blk = tid * 2 + i;                 // 512 blocks of 8 floats
      int m = blk >> 5, k0 = (blk & 31) << 3;
      const float4* xp4 = (const float4*)&s_xf[buf][m * 256 + k0];
      float4 a = xp4[0], b = xp4[1];
      float v[8] = {a.x, a.y, a.z, a.w, b.x, b.y, b.z, b.w};
      *(uint4*)&s_xp[pack_off(m, k0)] = pack8(v);
    }
    __syncthreads();

    v8f gi[3], gh[3];
#pragma unroll
    for (int j = 0; j < 3; ++j)
#pragma unroll
      for (int r = 0; r < 8; ++r) { gi[j][r] = 0.0f; gh[j][r] = 0.0f; }

    // ---- gi = x @ Wih^T  (K = 256) -----------------------------------------
#pragma unroll
    for (int kt = 0; kt < 8; ++kt) {
      FragU A;
      const u16* ap = &s_xp[(kt * 32 + lane) * 16];
      A.q[0] = *(const uint4*)ap;
      A.q[1] = *(const uint4*)(ap + 8);
#pragma unroll
      for (int j = 0; j < 3; ++j) {
        int nt = wv * 3 + j;
        const u16* bp = W1p + (((size_t)nt * 8 + kt) * 32 + lane) * 16;
        FragU B;
        B.q[0] = *(const uint4*)bp;
        B.q[1] = *(const uint4*)(bp + 8);
        gi[j] = wmma_bf16(A, B, gi[j]);
      }
    }
    // ---- gh = h @ Whh^T  (K = 128) -----------------------------------------
#pragma unroll
    for (int kt = 0; kt < 4; ++kt) {
      FragU A;
      const u16* ap = &s_hp[(kt * 32 + lane) * 16];
      A.q[0] = *(const uint4*)ap;
      A.q[1] = *(const uint4*)(ap + 8);
#pragma unroll
      for (int j = 0; j < 3; ++j) {
        int nt = wv * 3 + j;
        const u16* bp = W2p + (((size_t)nt * 4 + kt) * 32 + lane) * 16;
        FragU B;
        B.q[0] = *(const uint4*)bp;
        B.q[1] = *(const uint4*)(bp + 8);
        gh[j] = wmma_bf16(A, B, gh[j]);
      }
    }
    // ---- spill accumulators to LDS for cross-wave gate recombination -------
#pragma unroll
    for (int j = 0; j < 3; ++j) {
      int cb = (wv * 3 + j) * 16 + col0;
#pragma unroll
      for (int r = 0; r < 8; ++r) {
        s_gi[(mrow + r) * GSTR + cb] = gi[j][r];
        s_gh[(mrow + r) * GSTR + cb] = gh[j][r];
      }
    }
    __syncthreads();

    // ---- GRU nonlinearity epilogue; update h pack --------------------------
    {
      int m = tid >> 4, j0 = (tid & 15) << 3;
      uint4 hold = *(const uint4*)&s_hp[pack_off(m, j0)];
      unsigned hw[4] = {hold.x, hold.y, hold.z, hold.w};
      const float* gim = &s_gi[m * GSTR];
      const float* ghm = &s_gh[m * GSTR];
      float nv[8];
#pragma unroll
      for (int e = 0; e < 8; ++e) {
        int j = j0 + e;
        float hv = bf2f((hw[e >> 1] >> ((e & 1) * 16)) & 0xFFFFu);
        float r = sigf(gim[j] + s_bi[j] + ghm[j] + s_bh[j]);
        float z = sigf(gim[128 + j] + s_bi[128 + j] + ghm[128 + j] + s_bh[128 + j]);
        float hn = ghm[256 + j] + s_bh[256 + j];
        float nn = tanhf_fast(gim[256 + j] + s_bi[256 + j] + r * hn);
        nv[e] = (1.0f - z) * nn + z * hv;
      }
      uint4 np = pack8(nv);
      *(uint4*)&s_hp[pack_off(m, j0)] = np;
      if (c == 7)
        *(uint4*)&hc_pack[((size_t)(t * 256 + ntile)) * 2048 + pack_off(m, j0)] = np;
    }
    __syncthreads();
  }
  // drain the speculative last prefetch before the wave exits
  asm volatile("s_wait_asynccnt 0x0" ::: "memory");
}

// ---------------------------------------------------------------------------
// Stage 2a: user GRU gates for one time step.  Grid 256 groups (16 rows each).
// gi = h_c @ Wih_u[:, :128]^T + prev_y * wy + bih_u ;  gh = h @ Whh_u^T + bhh_u
// ---------------------------------------------------------------------------
__global__ __launch_bounds__(256, 1) void user_gates_kernel(
    const u16* __restrict__ hcp_t, const u16* __restrict__ hup,
    const u16* __restrict__ W3p, const u16* __restrict__ W4p,
    const float* __restrict__ wy, const float* __restrict__ bih,
    const float* __restrict__ bhh, const float* __restrict__ py, int py_valid,
    float* __restrict__ hg_f32, u16* __restrict__ hgp) {
  __shared__ float s_gi[16 * GSTR];
  __shared__ float s_gh[16 * GSTR];
  const int tid = threadIdx.x, lane = tid & 31, wv = tid >> 5;
  const int ntile = blockIdx.x;
  const int mrow = (lane >> 4) << 3, col0 = lane & 15;
  const u16* Ac = hcp_t + (size_t)ntile * 2048;
  const u16* Au = hup + (size_t)ntile * 2048;

  v8f gi[3], gh[3];
#pragma unroll
  for (int j = 0; j < 3; ++j)
#pragma unroll
    for (int r = 0; r < 8; ++r) { gi[j][r] = 0.0f; gh[j][r] = 0.0f; }

#pragma unroll
  for (int kt = 0; kt < 4; ++kt) {
    FragU A1, A2;
    const u16* a1 = Ac + (kt * 32 + lane) * 16;
    const u16* a2 = Au + (kt * 32 + lane) * 16;
    A1.q[0] = *(const uint4*)a1; A1.q[1] = *(const uint4*)(a1 + 8);
    A2.q[0] = *(const uint4*)a2; A2.q[1] = *(const uint4*)(a2 + 8);
#pragma unroll
    for (int j = 0; j < 3; ++j) {
      int nt = wv * 3 + j;
      const u16* b3 = W3p + (((size_t)nt * 4 + kt) * 32 + lane) * 16;
      const u16* b4 = W4p + (((size_t)nt * 4 + kt) * 32 + lane) * 16;
      FragU B3, B4;
      B3.q[0] = *(const uint4*)b3; B3.q[1] = *(const uint4*)(b3 + 8);
      B4.q[0] = *(const uint4*)b4; B4.q[1] = *(const uint4*)(b4 + 8);
      gi[j] = wmma_bf16(A1, B3, gi[j]);
      gh[j] = wmma_bf16(A2, B4, gh[j]);
    }
  }
#pragma unroll
  for (int j = 0; j < 3; ++j) {
    int cb = (wv * 3 + j) * 16 + col0;
#pragma unroll
    for (int r = 0; r < 8; ++r) {
      s_gi[(mrow + r) * GSTR + cb] = gi[j][r];
      s_gh[(mrow + r) * GSTR + cb] = gh[j][r];
    }
  }
  __syncthreads();

  {
    int m = tid >> 4, j0 = (tid & 15) << 3;
    int n = ntile * 16 + m;
    float pv = py_valid ? py[n] : 0.0f;
    uint4 hold = *(const uint4*)&Au[pack_off(m, j0)];
    unsigned hw[4] = {hold.x, hold.y, hold.z, hold.w};
    const float* gim = &s_gi[m * GSTR];
    const float* ghm = &s_gh[m * GSTR];
    float nv[8];
#pragma unroll
    for (int e = 0; e < 8; ++e) {
      int j = j0 + e;
      float hv = bf2f((hw[e >> 1] >> ((e & 1) * 16)) & 0xFFFFu);
      float ir = gim[j] + bih[j] + pv * wy[j];
      float iz = gim[128 + j] + bih[128 + j] + pv * wy[128 + j];
      float in_ = gim[256 + j] + bih[256 + j] + pv * wy[256 + j];
      float r = sigf(ir + ghm[j] + bhh[j]);
      float z = sigf(iz + ghm[128 + j] + bhh[128 + j]);
      float hn = ghm[256 + j] + bhh[256 + j];
      float nn = tanhf_fast(in_ + r * hn);
      float hnew = (1.0f - z) * nn + z * hv;
      hg_f32[(size_t)n * 128 + j] = hnew;
      nv[e] = hnew;
    }
    *(uint4*)&hgp[(size_t)ntile * 2048 + pack_off(m, j0)] = pack8(nv);
  }
}

// ---------------------------------------------------------------------------
// Stage 2b: GraphSAGE + classifier for one time step.  8 waves x 1 col tile.
// ---------------------------------------------------------------------------
__global__ __launch_bounds__(256, 1) void sage_kernel(
    const u16* __restrict__ hgp, const float* __restrict__ neigh,
    const float* __restrict__ deg, const u16* __restrict__ W5p,
    const u16* __restrict__ W6p, const float* __restrict__ bs,
    const float* __restrict__ bn, const float* __restrict__ Wc,
    const float* __restrict__ bc, u16* __restrict__ hup,
    float* __restrict__ out_logits) {
  __shared__ __align__(16) u16 s_np[4 * 512];   // neigh mean, A-frag order
  __shared__ float s_h[16 * 136];
  __shared__ float s_red[256];
  const int tid = threadIdx.x, lane = tid & 31, wv = tid >> 5;
  const int ntile = blockIdx.x;
  const int mrow = (lane >> 4) << 3, col0 = lane & 15;

  {  // build bf16 A pack of neighbor mean
    int m = tid >> 4, j0 = (tid & 15) << 3;
    int n = ntile * 16 + m;
    float d = deg[n];
    float inv = 1.0f / fmaxf(d, 1.0f);
    float v[8];
#pragma unroll
    for (int e = 0; e < 8; ++e) v[e] = neigh[(size_t)n * 128 + j0 + e] * inv;
    *(uint4*)&s_np[pack_off(m, j0)] = pack8(v);
  }
  __syncthreads();

  v8f a_self, a_nb;
#pragma unroll
  for (int r = 0; r < 8; ++r) { a_self[r] = 0.0f; a_nb[r] = 0.0f; }

  const u16* Ag = hgp + (size_t)ntile * 2048;
#pragma unroll
  for (int kt = 0; kt < 4; ++kt) {
    FragU A1, A2;
    const u16* a1 = Ag + (kt * 32 + lane) * 16;
    const u16* a2 = &s_np[(kt * 32 + lane) * 16];
    A1.q[0] = *(const uint4*)a1; A1.q[1] = *(const uint4*)(a1 + 8);
    A2.q[0] = *(const uint4*)a2; A2.q[1] = *(const uint4*)(a2 + 8);
    const u16* b5 = W5p + (((size_t)wv * 4 + kt) * 32 + lane) * 16;
    const u16* b6 = W6p + (((size_t)wv * 4 + kt) * 32 + lane) * 16;
    FragU B5, B6;
    B5.q[0] = *(const uint4*)b5; B5.q[1] = *(const uint4*)(b5 + 8);
    B6.q[0] = *(const uint4*)b6; B6.q[1] = *(const uint4*)(b6 + 8);
    a_self = wmma_bf16(A1, B5, a_self);
    a_nb = wmma_bf16(A2, B6, a_nb);
  }

  {  // relu(self + bs + has_neigh*(neigh + bn))
    int col = wv * 16 + col0;
#pragma unroll
    for (int r = 0; r < 8; ++r) {
      int m = mrow + r;
      int n = ntile * 16 + m;
      bool hasn = deg[n] > 0.0f;
      float v = a_self[r] + bs[col];
      if (hasn) v += a_nb[r] + bn[col];
      s_h[m * 136 + col] = fmaxf(v, 0.0f);
    }
  }
  __syncthreads();

  {  // repack h_new for next step + classifier partial dot
    int m = tid >> 4, j0 = (tid & 15) << 3;
    float v[8];
    float p = 0.0f;
#pragma unroll
    for (int e = 0; e < 8; ++e) {
      v[e] = s_h[m * 136 + j0 + e];
      p += v[e] * Wc[j0 + e];
    }
    *(uint4*)&hup[(size_t)ntile * 2048 + pack_off(m, j0)] = pack8(v);
    s_red[tid] = p;
  }
  __syncthreads();
  if (tid < 16) {
    float s = 0.0f;
#pragma unroll
    for (int q = 0; q < 16; ++q) s += s_red[tid * 16 + q];
    out_logits[ntile * 16 + tid] = s + bc[0];
  }
}

// ---------------------------------------------------------------------------
extern "C" void kernel_launch(void* const* d_in, const int* in_sizes, int n_in,
                              void* d_out, int out_size, void* d_ws, size_t ws_size,
                              hipStream_t stream) {
  (void)in_sizes; (void)n_in; (void)out_size; (void)ws_size;
  const float* comments = (const float*)d_in[0];
  const float* truth    = (const float*)d_in[1];
  const int*   esrc     = (const int*)d_in[2];
  const int*   edst     = (const int*)d_in[3];
  const float* Wih_c    = (const float*)d_in[4];
  const float* Whh_c    = (const float*)d_in[5];
  const float* bih_c    = (const float*)d_in[6];
  const float* bhh_c    = (const float*)d_in[7];
  const float* Wih_u    = (const float*)d_in[8];
  const float* Whh_u    = (const float*)d_in[9];
  const float* bih_u    = (const float*)d_in[10];
  const float* bhh_u    = (const float*)d_in[11];
  const float* Ws_      = (const float*)d_in[12];
  const float* bs_      = (const float*)d_in[13];
  const float* Wn_      = (const float*)d_in[14];
  const float* bn_      = (const float*)d_in[15];
  const float* Wc_      = (const float*)d_in[16];
  const float* bc_      = (const float*)d_in[17];

  char* ws = (char*)d_ws;
  size_t off = 0;
  auto take = [&](size_t bytes) -> void* {
    off = (off + 255) & ~(size_t)255;
    void* r = ws + off;
    off += bytes;
    return r;
  };
  u16*   W1p = (u16*)take((size_t)24 * 8 * 512 * 2);
  u16*   W2p = (u16*)take((size_t)24 * 4 * 512 * 2);
  u16*   W3p = (u16*)take((size_t)24 * 4 * 512 * 2);
  u16*   W4p = (u16*)take((size_t)24 * 4 * 512 * 2);
  u16*   W5p = (u16*)take((size_t)8 * 4 * 512 * 2);
  u16*   W6p = (u16*)take((size_t)8 * 4 * 512 * 2);
  float* wy  = (float*)take(384 * 4);
  u16*   hcp = (u16*)take((size_t)16 * 256 * 2048 * 2);
  u16*   hup = (u16*)take((size_t)256 * 2048 * 2);
  u16*   hgp = (u16*)take((size_t)256 * 2048 * 2);
  float* hgf = (float*)take((size_t)4096 * 128 * 4);
  float* nacc = (float*)take((size_t)4096 * 128 * 4);
  float* deg = (float*)take(4096 * 4);
  float* logits = (float*)d_out;

  // weight packing (once per call, deterministic)
  pack_w_kernel<<<48, 256, 0, stream>>>(Wih_c, W1p, 384, 256, 256);
  pack_w_kernel<<<24, 256, 0, stream>>>(Whh_c, W2p, 384, 128, 128);
  pack_w_kernel<<<24, 256, 0, stream>>>(Wih_u, W3p, 384, 128, 129);
  pack_w_kernel<<<24, 256, 0, stream>>>(Whh_u, W4p, 384, 128, 128);
  pack_w_kernel<<<8, 256, 0, stream>>>(Ws_, W5p, 128, 128, 128);
  pack_w_kernel<<<8, 256, 0, stream>>>(Wn_, W6p, 128, 128, 128);
  copy_wy_kernel<<<2, 256, 0, stream>>>(Wih_u, wy);

  zero_u16_kernel<<<2048, 256, 0, stream>>>(hup, 256 * 2048);
  zero_f32_kernel<<<16, 256, 0, stream>>>(deg, 4096);
  deg_kernel<<<256, 256, 0, stream>>>(esrc, deg, 65536);

  // Stage 1: comment encoder (dominant GEMM work)
  comment_gru_kernel<<<4096, 256, 0, stream>>>(comments, W1p, W2p, bih_c, bhh_c, hcp);

  // Stage 2: sequential time steps
  for (int t = 0; t < 16; ++t) {
    zero_f32_kernel<<<2048, 256, 0, stream>>>(nacc, 4096 * 128);
    const float* py = (t > 0) ? (truth + (size_t)(t - 1) * 4096) : truth;
    user_gates_kernel<<<256, 256, 0, stream>>>(hcp + (size_t)t * 256 * 2048, hup, W3p,
                                               W4p, wy, bih_u, bhh_u, py, t > 0 ? 1 : 0,
                                               hgf, hgp);
    scatter_kernel<<<32768, 256, 0, stream>>>(esrc, edst, hgf, nacc, 65536);
    sage_kernel<<<256, 256, 0, stream>>>(hgp, nacc, deg, W5p, W6p, bs_, bn_, Wc_, bc_,
                                         hup, logits + (size_t)t * 4096);
  }
  copy_truth_kernel<<<256, 256, 0, stream>>>(truth, logits + 65536, 65536);
}
